// DIAMNet_82935818486377
// MI455X (gfx1250) — matrix operand order repaired and checked by reading, not compiled, not run on hardware
//
#include <hip/hip_runtime.h>

typedef __bf16 bf16;
typedef __attribute__((ext_vector_type(16))) __bf16 v16bf;
typedef __attribute__((ext_vector_type(8)))  __bf16 v8bf;
typedef __attribute__((ext_vector_type(8)))  float  v8f;
typedef __attribute__((ext_vector_type(4)))  float  v4f;

#define DEVINL static __device__ __forceinline__

// ---------------------------------------------------------------------------
// WMMA helpers (CDNA5 gfx1250: D = A(16x32 bf16) x B(32x16 bf16) + C(16x16 f32))
// ---------------------------------------------------------------------------
DEVINL v8f wmma_bf16(v16bf a, v16bf b, v8f c) {
  return __builtin_amdgcn_wmma_f32_16x16x32_bf16(false, a, false, b, (short)0, c,
                                                 false, false);
}

// A-fragment: lane L holds A[m][kb..kb+15], kb=(L<16?0:16)
DEVINL v16bf afrag_bf(const bf16* row, int kb) {
  v8bf lo = *(const v8bf*)(row + kb);
  v8bf hi = *(const v8bf*)(row + kb + 8);
  v16bf r;
#pragma unroll
  for (int i = 0; i < 8; ++i) { r[i] = lo[i]; r[i + 8] = hi[i]; }
  return r;
}

DEVINL v16bf afrag_f32(const float* row, int kb) {
  v4f a0 = *(const v4f*)(row + kb);
  v4f a1 = *(const v4f*)(row + kb + 4);
  v4f a2 = *(const v4f*)(row + kb + 8);
  v4f a3 = *(const v4f*)(row + kb + 12);
  v16bf r;
#pragma unroll
  for (int i = 0; i < 4; ++i) {
    r[i]      = (bf16)a0[i];
    r[i + 4]  = (bf16)a1[i];
    r[i + 8]  = (bf16)a2[i];
    r[i + 12] = (bf16)a3[i];
  }
  return r;
}

// B-fragment: lane L holds 16 contiguous K values of column n=L%16, at K-offset
// kb2=(L<16?0:16). Caller passes the already-offset pointer.
DEVINL v16bf bfrag16(const bf16* p) {
  v8bf lo = *(const v8bf*)p;
  v8bf hi = *(const v8bf*)(p + 8);
  v16bf r;
#pragma unroll
  for (int i = 0; i < 8; ++i) { r[i] = lo[i]; r[i + 8] = hi[i]; }
  return r;
}

// ---------------------------------------------------------------------------
// Generic WMMA GEMM:  C[M,N] = act(A[M,K] @ W[N,K]^T + bias)
//   - A is f32 (converted on the fly) or bf16, row stride lda
//   - W bf16 [N,K] row-major, double-buffered 64x32 LDS tiles (padded rows,
//     conflict-free ds_load_b128); ONE barrier per K-step
//   - all 4 B-fragments loaded before the 4 WMMAs (graduated DScnt waits)
//   - epilogues: f32 C, bf16 C (with column offset, for concat buffers),
//     and per-(batch,head) transposed bf16 C ([B,H,64,tL]) for attention V
//   - block: 256 threads = 8 waves; wave tile 16(M) x 64(N); block 128 x 64
// ---------------------------------------------------------------------------
__global__ __launch_bounds__(256) void wmma_gemm_kernel(
    const void* __restrict__ A, int a_is_f32, long lda,
    const bf16* __restrict__ W, const float* __restrict__ bias,
    int M, int N, int K,
    float* __restrict__ Cf, long ldc,
    bf16* __restrict__ Cb, long ldcb, long cb_off,
    bf16* __restrict__ CbT, int tL,
    int act) {
  __shared__ bf16 ldsW[2][64 * 40];  // 2 x (64 rows x 32 bf16 padded to 40)
  const int tid  = threadIdx.x;
  const int w    = tid >> 5;
  const int lane = tid & 31;
  const int ln   = lane & 15;
  const int half = lane >> 4;
  const int n0   = blockIdx.x * 64;
  const int m0   = blockIdx.y * 128 + w * 16;
  const bool wvalid = (m0 < M);           // M is always a multiple of 16 here
  const int mA   = wvalid ? m0 : 0;       // clamp so idle waves still load safely
  const int kbA  = half ? 16 : 0;
  const int kbB  = half ? 16 : 0;
  const int wr   = tid >> 2;              // cooperative W-tile row 0..63
  const int wc   = (tid & 3) * 8;         // 8-element chunk within row

  const bf16*  Ab   = (const bf16*)A;
  const float* Af   = (const float*)A;
  const size_t arow = (size_t)(mA + ln) * (size_t)lda;
  const size_t wrow = (size_t)(n0 + wr) * (size_t)K + wc;

  v8f acc[4];
#pragma unroll
  for (int t = 0; t < 4; ++t)
#pragma unroll
    for (int r = 0; r < 8; ++r) acc[t][r] = 0.f;

  // stage first tile into buffer 0
  *(v8bf*)(&ldsW[0][wr * 40 + wc]) = *(const v8bf*)(W + wrow);
  __syncthreads();

  const int niter = K >> 5;
  for (int it = 0; it < niter; ++it) {
    const int k0  = it << 5;
    const int cur = it & 1;
    const int nxt = cur ^ 1;
    if (it + 1 < niter) {
      // stage next tile into the other buffer while current one feeds WMMAs
      *(v8bf*)(&ldsW[nxt][wr * 40 + wc]) = *(const v8bf*)(W + wrow + k0 + 32);
      if (it + 2 < niter)
        __builtin_prefetch(W + wrow + k0 + 64, 0, 3);
    }

    v16bf a = a_is_f32 ? afrag_f32(Af + arow + k0, kbA)
                       : afrag_bf (Ab + arow + k0, kbA);
    v16bf b[4];
#pragma unroll
    for (int t = 0; t < 4; ++t)
      b[t] = bfrag16(&ldsW[cur][(t * 16 + ln) * 40 + kbB]);
#pragma unroll
    for (int t = 0; t < 4; ++t)
      acc[t] = wmma_bf16(a, b[t], acc[t]);

    __syncthreads();  // cur fully consumed; next iter may overwrite it
  }

  // Epilogue. C layout: VGPR r -> row m0 + r + 8*half, column n0 + t*16 + ln.
  const int mrow0 = m0 + half * 8;
#pragma unroll
  for (int t = 0; t < 4; ++t) {
    const int n = n0 + t * 16 + ln;
    const float bv = bias ? bias[n] : 0.f;
    float vals[8];
#pragma unroll
    for (int r = 0; r < 8; ++r) {
      float v = acc[t][r] + bv;
      if (act == 1) v = (v < 0.f) ? v * 0.01f : v;  // leaky_relu(0.01)
      vals[r] = v;
    }
    if (wvalid) {
      if (Cf) {
#pragma unroll
        for (int r = 0; r < 8; ++r)
          Cf[(size_t)(mrow0 + r) * (size_t)ldc + n] = vals[r];
      }
      if (Cb) {
#pragma unroll
        for (int r = 0; r < 8; ++r)
          Cb[(size_t)(mrow0 + r) * (size_t)ldcb + cb_off + n] = (bf16)vals[r];
      }
      if (CbT) {  // transposed store: [B, H=8, d=64, tL]; 8 contiguous bf16/lane
        const int bb = m0 / tL;
        const int j0 = (m0 % tL) + half * 8;
        const int hh = n >> 6, dd = n & 63;
        v8bf pk;
#pragma unroll
        for (int r = 0; r < 8; ++r) pk[r] = (bf16)vals[r];
        *(v8bf*)(CbT + (((size_t)bb * 8 + hh) * 64 + dd) * (size_t)tL + j0) = pk;
      }
    }
  }
}

// ---------------------------------------------------------------------------
// Flash attention per (batch, head): 64 queries, D=64, key length L, mask j<klen.
// Block = 128 threads = 4 waves, wave owns 16 query rows. Online softmax with
// width-16 shfl reductions; P routed through padded LDS to become an A-fragment.
// hq/hk: [B, tokens, 512] bf16 ; hvT: [B, H, 64, L] bf16 ; out: [B, 64, 512] bf16
// ---------------------------------------------------------------------------
__global__ __launch_bounds__(128) void attn_kernel(
    const bf16* __restrict__ hq, const bf16* __restrict__ hk,
    const bf16* __restrict__ hvT, bf16* __restrict__ outv,
    const int* __restrict__ lens, int L) {
  __shared__ bf16 ldsP[4][16 * 40];
  const int bh = blockIdx.x, b = bh >> 3, h = bh & 7;
  const int tid = threadIdx.x, w = tid >> 5, lane = tid & 31;
  const int ln = lane & 15, half = lane >> 4;
  const int klen = lens[b];
  const int q0 = w * 16;
  const int kbA = half ? 16 : 0;
  const int kbB = half ? 16 : 0;

  const bf16* qrow = hq + ((size_t)b * 64 + q0 + ln) * 512 + h * 64;
  v16bf aQ0 = afrag_bf(qrow, kbA);        // d 0..31
  v16bf aQ1 = afrag_bf(qrow + 32, kbA);   // d 32..63

  const bf16* kbase = hk + (size_t)b * L * 512 + h * 64;
  const bf16* vbase = hvT + (size_t)bh * 64 * (size_t)L;

  float rm[8], rs[8];
  v8f o[4];
#pragma unroll
  for (int r = 0; r < 8; ++r) { rm[r] = -1e30f; rs[r] = 0.f; }
#pragma unroll
  for (int t = 0; t < 4; ++t)
#pragma unroll
    for (int r = 0; r < 8; ++r) o[t][r] = 0.f;

  for (int kt = 0; kt < L; kt += 32) {
    // ---- scores: load all 4 K-fragments, then 4 WMMAs ----
    v16bf bk[4];
#pragma unroll
    for (int sub = 0; sub < 2; ++sub) {
      const bf16* kp = kbase + (size_t)(kt + sub * 16 + ln) * 512;
      bk[2 * sub]     = bfrag16(kp + kbB);        // d 0..31
      bk[2 * sub + 1] = bfrag16(kp + 32 + kbB);   // d 32..63
    }
    if (kt + 32 < L)
      __builtin_prefetch(kbase + (size_t)(kt + 32 + ln) * 512, 0, 3);

    v8f s01[2];
#pragma unroll
    for (int sub = 0; sub < 2; ++sub) {
      v8f s;
#pragma unroll
      for (int r = 0; r < 8; ++r) s[r] = 0.f;
      s = wmma_bf16(aQ0, bk[2 * sub], s);
      s = wmma_bf16(aQ1, bk[2 * sub + 1], s);
      const int j = kt + sub * 16 + ln;  // this lane's key column
      const bool valid = (j < klen);
#pragma unroll
      for (int r = 0; r < 8; ++r)
        s[r] = valid ? s[r] * 0.125f : -1e30f;  // scale 1/sqrt(64), mask
      s01[sub] = s;
    }

    // ---- online softmax over this 32-key block ----
    float alpha[8];
#pragma unroll
    for (int r = 0; r < 8; ++r) {
      float s0 = s01[0][r], s1 = s01[1][r];
      float t = fmaxf(s0, s1);
      t = fmaxf(t, __shfl_xor(t, 1, 16));
      t = fmaxf(t, __shfl_xor(t, 2, 16));
      t = fmaxf(t, __shfl_xor(t, 4, 16));
      t = fmaxf(t, __shfl_xor(t, 8, 16));
      const float nm = fmaxf(rm[r], t);
      const float al = __expf(rm[r] - nm);
      const float p0 = __expf(s0 - nm);
      const float p1 = __expf(s1 - nm);
      float ps = p0 + p1;
      ps += __shfl_xor(ps, 1, 16);
      ps += __shfl_xor(ps, 2, 16);
      ps += __shfl_xor(ps, 4, 16);
      ps += __shfl_xor(ps, 8, 16);
      rs[r] = rs[r] * al + ps;
      rm[r] = nm;
      alpha[r] = al;
      const int row = r + half * 8;                 // P row (query)
      ldsP[w][row * 40 + ln]      = (bf16)p0;       // keys kt..kt+15
      ldsP[w][row * 40 + 16 + ln] = (bf16)p1;       // keys kt+16..kt+31
    }
#pragma unroll
    for (int t = 0; t < 4; ++t)
#pragma unroll
      for (int r = 0; r < 8; ++r) o[t][r] *= alpha[r];

    // ---- P x V: P as A-fragment (row = ln, 32 keys); V^T keys-contiguous ----
    v16bf aP = afrag_bf(&ldsP[w][ln * 40], kbA);
    v16bf bv[4];
#pragma unroll
    for (int t = 0; t < 4; ++t)
      bv[t] = bfrag16(vbase + (size_t)(t * 16 + ln) * L + kt + kbB);
#pragma unroll
    for (int t = 0; t < 4; ++t)
      o[t] = wmma_bf16(aP, bv[t], o[t]);
  }

#pragma unroll
  for (int t = 0; t < 4; ++t)
#pragma unroll
    for (int r = 0; r < 8; ++r) {
      const int m = q0 + r + half * 8;
      const int n = h * 64 + t * 16 + ln;
      outv[((size_t)b * 64 + m) * 512 + n] = (bf16)(o[t][r] / rs[r]);
    }
}

// ---------------------------------------------------------------------------
// Small helper kernels
// ---------------------------------------------------------------------------
__global__ void cvt_kernel(const float* __restrict__ s, bf16* __restrict__ d, long n) {
  long i = (long)blockIdx.x * blockDim.x + threadIdx.x;
  if (i < n) d[i] = (bf16)s[i];
}

// masked mean over MEM=64 chunks of 64 graph rows -> pooled bf16 [B*64, 512]
__global__ __launch_bounds__(256) void pool_kernel(const float* __restrict__ graph,
                                                   const int* __restrict__ glen,
                                                   bf16* __restrict__ pooled) {
  const int bm = blockIdx.x;
  const int b = bm >> 6, m = bm & 63;
  int cnt = glen[b] - m * 64;
  cnt = cnt < 0 ? 0 : (cnt > 64 ? 64 : cnt);
  const float denom = (float)(cnt < 1 ? 1 : cnt);
  const float* base = graph + ((size_t)b * 4096 + m * 64) * 512;
  for (int d = threadIdx.x; d < 512; d += 256) {
    float s = 0.f;
    for (int i = 0; i < cnt; ++i) s += base[(size_t)i * 512 + d];
    pooled[((size_t)b * 64 + m) * 512 + d] = (bf16)(s / denom);
  }
}

__global__ void seed_qcat_kernel(const bf16* __restrict__ memc, bf16* __restrict__ qcat) {
  long i = (long)blockIdx.x * blockDim.x + threadIdx.x;
  if (i < 2048L * 512) qcat[(i >> 9) * 1024 + (i & 511)] = memc[i];
}

// mem = sig(z)*mem + (1-sig(z))*attn_out ; refresh bf16 copies (memc + qcat[:, :512])
__global__ void blend_kernel(const float* __restrict__ z, const float* __restrict__ aout,
                             float* __restrict__ memf, bf16* __restrict__ memc,
                             bf16* __restrict__ qcat) {
  long i = (long)blockIdx.x * blockDim.x + threadIdx.x;
  if (i < 2048L * 512) {
    const float g = 1.f / (1.f + __expf(-z[i]));
    const float v = g * memf[i] + (1.f - g) * aout[i];
    memf[i] = v;
    const bf16 bv = (bf16)v;
    memc[i] = bv;
    qcat[(i >> 9) * 1024 + (i & 511)] = bv;
  }
}

// y[b] = h[b,:] . W2 + b2
__global__ __launch_bounds__(256) void head_kernel(const float* __restrict__ h,
                                                   const float* __restrict__ W2,
                                                   const float* __restrict__ b2,
                                                   float* __restrict__ y) {
  __shared__ float red[8];
  const int b = blockIdx.x;
  float s = 0.f;
  for (int d = threadIdx.x; d < 512; d += 256) s += h[b * 512 + d] * W2[d];
#pragma unroll
  for (int off = 16; off > 0; off >>= 1) s += __shfl_xor(s, off);
  if ((threadIdx.x & 31) == 0) red[threadIdx.x >> 5] = s;
  __syncthreads();
  if (threadIdx.x == 0) {
    float t = 0.f;
#pragma unroll
    for (int i = 0; i < 8; ++i) t += red[i];
    y[b] = t + b2[0];
  }
}

// ---------------------------------------------------------------------------
// Host orchestration
// ---------------------------------------------------------------------------
static inline void launch_gemm(hipStream_t st, const void* A, int a_is_f32, long lda,
                               const bf16* W, const float* bias, int M, int N, int K,
                               float* Cf, long ldc, bf16* Cb, long ldcb, long cb_off,
                               bf16* CbT, int tL, int act) {
  dim3 grid((unsigned)(N / 64), (unsigned)((M + 127) / 128));
  wmma_gemm_kernel<<<grid, 256, 0, st>>>(A, a_is_f32, lda, W, bias, M, N, K,
                                         Cf, ldc, Cb, ldcb, cb_off, CbT, tL, act);
}

extern "C" void kernel_launch(void* const* d_in, const int* in_sizes, int n_in,
                              void* d_out, int out_size, void* d_ws, size_t ws_size,
                              hipStream_t stream) {
  (void)in_sizes; (void)n_in; (void)out_size; (void)ws_size;
  const float* pattern = (const float*)d_in[0];
  const float* graph   = (const float*)d_in[1];
  const int*   plen    = (const int*)d_in[2];
  const int*   glen    = (const int*)d_in[3];
  const float* Wp   = (const float*)d_in[4];  const float* bp   = (const float*)d_in[5];
  const float* Wg   = (const float*)d_in[6];  const float* bg   = (const float*)d_in[7];
  const float* Wm   = (const float*)d_in[8];  const float* bm   = (const float*)d_in[9];
  const float* pWq  = (const float*)d_in[10]; const float* pWk  = (const float*)d_in[11];
  const float* pWv  = (const float*)d_in[12]; const float* pWo  = (const float*)d_in[13];
  const float* pWgn = (const float*)d_in[14]; const float* pbgn = (const float*)d_in[15];
  const float* gWq  = (const float*)d_in[16]; const float* gWk  = (const float*)d_in[17];
  const float* gWv  = (const float*)d_in[18]; const float* gWo  = (const float*)d_in[19];
  const float* gWgn = (const float*)d_in[20]; const float* gbgn = (const float*)d_in[21];
  const float* W1   = (const float*)d_in[22]; const float* b1   = (const float*)d_in[23];
  const float* W2   = (const float*)d_in[24]; const float* b2   = (const float*)d_in[25];
  float* y = (float*)d_out;

  const long NP = 4096;     // B*PLEN rows
  const long NG = 131072;   // B*GLEN rows
  const long NM = 2048;     // B*MEM rows
  const long W512 = 512L * 512, WGN = 512L * 1024, W1N = 512L * 32768;

  char* ws = (char*)d_ws;
  size_t off = 0;
  auto alloc = [&](size_t bytes) -> char* {
    char* p = ws + off;
    off = (off + bytes + 255) & ~(size_t)255;
    return p;
  };
  bf16* wbWp   = (bf16*)alloc(W512 * 2);
  bf16* wbWg   = (bf16*)alloc(W512 * 2);
  bf16* wbWm   = (bf16*)alloc(W512 * 2);
  bf16* wbpWq  = (bf16*)alloc(W512 * 2);
  bf16* wbpWk  = (bf16*)alloc(W512 * 2);
  bf16* wbpWv  = (bf16*)alloc(W512 * 2);
  bf16* wbpWo  = (bf16*)alloc(W512 * 2);
  bf16* wbpWgn = (bf16*)alloc(WGN * 2);
  bf16* wbgWq  = (bf16*)alloc(W512 * 2);
  bf16* wbgWk  = (bf16*)alloc(W512 * 2);
  bf16* wbgWv  = (bf16*)alloc(W512 * 2);
  bf16* wbgWo  = (bf16*)alloc(W512 * 2);
  bf16* wbgWgn = (bf16*)alloc(WGN * 2);
  bf16* wbW1   = (bf16*)alloc(W1N * 2);
  bf16* p_bf   = (bf16*)alloc(NP * 512 * 2);
  bf16* g_bf   = (bf16*)alloc(NG * 512 * 2);
  bf16* hkp    = (bf16*)alloc(NP * 512 * 2);
  bf16* hvpT   = (bf16*)alloc(NP * 512 * 2);
  bf16* hkg    = (bf16*)alloc(NG * 512 * 2);
  bf16* hvgT   = (bf16*)alloc(NG * 512 * 2);
  bf16* pooled = (bf16*)alloc(NM * 512 * 2);
  float* memf  = (float*)alloc(NM * 512 * 4);
  bf16* memc   = (bf16*)alloc(NM * 512 * 2);
  bf16* qcat   = (bf16*)alloc(NM * 1024 * 2);
  bf16* hqb    = (bf16*)alloc(NM * 512 * 2);
  bf16* vecb   = (bf16*)alloc(NM * 512 * 2);
  float* aout  = (float*)alloc(NM * 512 * 4);
  float* zf    = (float*)alloc(NM * 512 * 4);
  float* hf    = (float*)alloc(32L * 512 * 4);

  auto CVT = [&](const float* s, bf16* d, long n) {
    cvt_kernel<<<dim3((unsigned)((n + 255) / 256)), 256, 0, stream>>>(s, d, n);
  };
  CVT(Wp, wbWp, W512);   CVT(Wg, wbWg, W512);   CVT(Wm, wbWm, W512);
  CVT(pWq, wbpWq, W512); CVT(pWk, wbpWk, W512); CVT(pWv, wbpWv, W512);
  CVT(pWo, wbpWo, W512); CVT(pWgn, wbpWgn, WGN);
  CVT(gWq, wbgWq, W512); CVT(gWk, wbgWk, W512); CVT(gWv, wbgWv, W512);
  CVT(gWo, wbgWo, W512); CVT(gWgn, wbgWgn, WGN);
  CVT(W1, wbW1, W1N);

  // mem_init pooling on raw graph
  pool_kernel<<<2048, 256, 0, stream>>>(graph, glen, pooled);

  // p = lrelu(pattern@Wp.T+bp), g = lrelu(graph@Wg.T+bg)  (bf16 only, A f32 fused-cvt)
  launch_gemm(stream, pattern, 1, 512, wbWp, bp, (int)NP, 512, 512,
              nullptr, 0, p_bf, 512, 0, nullptr, 0, 1);
  launch_gemm(stream, graph, 1, 512, wbWg, bg, (int)NG, 512, 512,
              nullptr, 0, g_bf, 512, 0, nullptr, 0, 1);

  // mem0 = lrelu(pooled@Wm.T+bm) -> f32 + bf16
  launch_gemm(stream, pooled, 0, 512, wbWm, bm, (int)NM, 512, 512,
              memf, 512, memc, 512, 0, nullptr, 0, 1);
  seed_qcat_kernel<<<4096, 256, 0, stream>>>(memc, qcat);

  // loop-invariant K/V projections (hoisted out of the 3 steps)
  launch_gemm(stream, p_bf, 0, 512, wbpWk, nullptr, (int)NP, 512, 512,
              nullptr, 0, hkp, 512, 0, nullptr, 0, 0);
  launch_gemm(stream, p_bf, 0, 512, wbpWv, nullptr, (int)NP, 512, 512,
              nullptr, 0, nullptr, 0, 0, hvpT, 128, 0);
  launch_gemm(stream, g_bf, 0, 512, wbgWk, nullptr, (int)NG, 512, 512,
              nullptr, 0, hkg, 512, 0, nullptr, 0, 0);
  launch_gemm(stream, g_bf, 0, 512, wbgWv, nullptr, (int)NG, 512, 512,
              nullptr, 0, nullptr, 0, 0, hvgT, 4096, 0);

  for (int step = 0; step < 3; ++step) {
    // --- pattern attention ---
    launch_gemm(stream, memc, 0, 512, wbpWq, nullptr, (int)NM, 512, 512,
                nullptr, 0, hqb, 512, 0, nullptr, 0, 0);
    attn_kernel<<<256, 128, 0, stream>>>(hqb, hkp, hvpT, vecb, plen, 128);
    launch_gemm(stream, vecb, 0, 512, wbpWo, nullptr, (int)NM, 512, 512,
                aout, 512, qcat, 1024, 512, nullptr, 0, 0);
    launch_gemm(stream, qcat, 0, 1024, wbpWgn, pbgn, (int)NM, 512, 1024,
                zf, 512, nullptr, 0, 0, nullptr, 0, 0);
    blend_kernel<<<4096, 256, 0, stream>>>(zf, aout, memf, memc, qcat);

    // --- graph attention ---
    launch_gemm(stream, memc, 0, 512, wbgWq, nullptr, (int)NM, 512, 512,
                nullptr, 0, hqb, 512, 0, nullptr, 0, 0);
    attn_kernel<<<256, 128, 0, stream>>>(hqb, hkg, hvgT, vecb, glen, 4096);
    launch_gemm(stream, vecb, 0, 512, wbgWo, nullptr, (int)NM, 512, 512,
                aout, 512, qcat, 1024, 512, nullptr, 0, 0);
    launch_gemm(stream, qcat, 0, 1024, wbgWgn, gbgn, (int)NM, 512, 1024,
                zf, 512, nullptr, 0, 0, nullptr, 0, 0);
    blend_kernel<<<4096, 256, 0, stream>>>(zf, aout, memf, memc, qcat);
  }

  // h = lrelu(mem.reshape(B, 32768) @ W1.T + b1) ; memc is contiguous [32, 32768]
  launch_gemm(stream, memc, 0, 32768, wbW1, b1, 32, 512, 32768,
              hf, 512, nullptr, 0, 0, nullptr, 0, 1);
  head_kernel<<<32, 256, 0, stream>>>(hf, W2, b2, y);
}
